// NILMFormer_2491081032089
// MI455X (gfx1250) — matrix operand order repaired and checked by reading, not compiled, not run on hardware
//
#include <hip/hip_runtime.h>
#include <math.h>

// ---------------- model dims ----------------
static const int NB    = 8;      // batch
static const int LL    = 2048;   // sequence length (conv domain)
static const int SS    = 2049;   // transformer tokens (L + stats token)
static const int DMODEL= 256;
static const int NH    = 8;
static const int HDIM  = 32;
static const int DFF   = 1024;
static const int CMAIN = 192;
static const int CEXO  = 64;
static const int CEMB  = 8;      // exogenous encoding channels
static const int BSROWS= NB * SS;    // 16392

// ---------------- WMMA types ----------------
typedef __attribute__((ext_vector_type(16))) __bf16 bf16x16;
typedef __attribute__((ext_vector_type(2)))  __bf16 bf16x2;
typedef __attribute__((ext_vector_type(8)))  float  f32x8;

union Frag {
  bf16x16 v;
  unsigned short u[16];
  uint4 q[2];
};

// hardware bf16 conversion (RNE) via fptrunc; clang lowers to v_cvt on gfx1250
__device__ __forceinline__ unsigned short f2bf(float f) {
  __bf16 h = (__bf16)f;
  return __builtin_bit_cast(unsigned short, h);
}

__device__ __forceinline__ unsigned int pk2(float a, float b) {
  bf16x2 h;
  h.x = (__bf16)a;
  h.y = (__bf16)b;
  return __builtin_bit_cast(unsigned int, h);
}

__device__ __forceinline__ float gelu_exact(float x) {
  return 0.5f * x * (1.0f + erff(x * 0.70710678118654752f));
}

// Fragment-order LDS addressing:
//  A-matrix (16x32, bf16): value (m,k) -> lane (k>>4)*16+m, elem k&15
//  B-matrix (32x16, bf16): value (kk,n) -> lane (kk>>4)*16+(n&15), elem kk&15

// =====================================================================
// Generic GEMM: C[m,n] = act( sum_k A[m,k]*W[n,k] + bias[n] ) + resid[m,n]
// A: MxK row-major f32, W: NxK row-major f32 ("oi" weights). K % 32 == 0.
// 128 threads = 4 waves; block tile 64x64; wave w -> rows [w*16, w*16+16).
// =====================================================================
__global__ void __launch_bounds__(128) gemm_wmma_kernel(
    const float* __restrict__ A, const float* __restrict__ W,
    const float* __restrict__ bias, const float* __restrict__ resid,
    float* __restrict__ C, int M, int N, int K, int act)
{
  __shared__ __align__(16) unsigned short AsF[4][32][16];
  __shared__ __align__(16) unsigned short WsF[4][32][16];

  const int tid  = threadIdx.x;
  const int wave = tid >> 5;
  const int lane = tid & 31;
  const int hi   = (lane >= 16) ? 1 : 0;
  const int lo   = lane & 15;
  const int m0   = blockIdx.x * 64;
  const int n0   = blockIdx.y * 64;

  f32x8 acc[4];
#pragma unroll
  for (int j = 0; j < 4; ++j)
#pragma unroll
    for (int r = 0; r < 8; ++r) acc[j][r] = 0.0f;

  for (int k0 = 0; k0 < K; k0 += 32) {
    // A tile 64x32 -> fragment order (float4 global loads, b64 LDS stores)
    for (int i = tid; i < 512; i += 128) {
      int r = i >> 3, c0 = (i & 7) * 4;
      int gm = m0 + r;
      float4 f = make_float4(0.f, 0.f, 0.f, 0.f);
      if (gm < M) f = *(const float4*)(A + (size_t)gm * K + k0 + c0);
      int sub = r >> 4, m = r & 15;
      int lane_d = (c0 >> 4) * 16 + m;
      int e = c0 & 15;
      uint2 p; p.x = pk2(f.x, f.y); p.y = pk2(f.z, f.w);
      *(uint2*)&AsF[sub][lane_d][e] = p;
    }
    // W tile 32x64 -> fragment order
    for (int i = tid; i < 512; i += 128) {
      int n = i >> 3, kk0 = (i & 7) * 4;
      int gn = n0 + n;
      float4 f = make_float4(0.f, 0.f, 0.f, 0.f);
      if (gn < N) f = *(const float4*)(W + (size_t)gn * K + k0 + kk0);
      int sub = n >> 4, nn = n & 15;
      int lane_d = (kk0 >> 4) * 16 + nn;
      int e = kk0 & 15;
      uint2 p; p.x = pk2(f.x, f.y); p.y = pk2(f.z, f.w);
      *(uint2*)&WsF[sub][lane_d][e] = p;
    }
    __syncthreads();

    Frag a;
    {
      const uint4* p = (const uint4*)&AsF[wave][lane][0];
      a.q[0] = p[0]; a.q[1] = p[1];
    }
#pragma unroll
    for (int j = 0; j < 4; ++j) {
      Frag bm;
      const uint4* p = (const uint4*)&WsF[j][lane][0];
      bm.q[0] = p[0]; bm.q[1] = p[1];
      acc[j] = __builtin_amdgcn_wmma_f32_16x16x32_bf16(
          false, a.v, false, bm.v, (short)0, acc[j], false, false);
    }
    __syncthreads();
  }

  int rbase = m0 + wave * 16 + hi * 8;
#pragma unroll
  for (int j = 0; j < 4; ++j) {
    int col = n0 + j * 16 + lo;
#pragma unroll
    for (int r = 0; r < 8; ++r) {
      int row = rbase + r;
      if (row < M && col < N) {
        float v = acc[j][r];
        if (bias) v += bias[col];
        if (act == 1) v = gelu_exact(v);
        if (resid) v += resid[(size_t)row * N + col];
        C[(size_t)row * N + col] = v;
      }
    }
  }
}

// =====================================================================
// Dilated conv1d (SAME) as implicit GEMM, fused bias + exact GELU.
// in: (B, CIN, L) f32; w: (CO, CIN, 3) f32; out: (B, CO, L) activated.
// Block tile: 64 positions x 64 channels; K = CIN*3 (multiple of 32).
// =====================================================================
__global__ void __launch_bounds__(128) conv_wmma_kernel(
    const float* __restrict__ in, const float* __restrict__ w,
    const float* __restrict__ bias, float* __restrict__ out,
    int CIN, int CO, int L, int dil)
{
  __shared__ __align__(16) unsigned short AsF[4][32][16];
  __shared__ __align__(16) unsigned short WsF[4][32][16];

  const int tid  = threadIdx.x;
  const int wave = tid >> 5;
  const int lane = tid & 31;
  const int hi   = (lane >= 16) ? 1 : 0;
  const int lo   = lane & 15;
  const int b    = blockIdx.z;
  const int t0   = blockIdx.x * 64;
  const int n0   = blockIdx.y * 64;
  const int K    = CIN * 3;
  const float* inb = in + (size_t)b * CIN * L;

  f32x8 acc[4];
#pragma unroll
  for (int j = 0; j < 4; ++j)
#pragma unroll
    for (int r = 0; r < 8; ++r) acc[j][r] = 0.0f;

  for (int k0 = 0; k0 < K; k0 += 32) {
    // gathered A tile (im2col on the fly)
    for (int i = tid; i < 512; i += 128) {
      int r = i >> 3, c0 = (i & 7) * 4;
      float f[4];
#pragma unroll
      for (int z = 0; z < 4; ++z) {
        int gk = k0 + c0 + z;
        float vv = 0.0f;
        if (gk < K) {
          int ci = gk / 3, ks = gk % 3;
          int tt = t0 + r + dil * (ks - 1);
          if (tt >= 0 && tt < L) vv = inb[(size_t)ci * L + tt];
        }
        f[z] = vv;
      }
      int sub = r >> 4, m = r & 15;
      int lane_d = (c0 >> 4) * 16 + m;
      int e = c0 & 15;
      uint2 p; p.x = pk2(f[0], f[1]); p.y = pk2(f[2], f[3]);
      *(uint2*)&AsF[sub][lane_d][e] = p;
    }
    for (int i = tid; i < 512; i += 128) {
      int n = i >> 3, kk0 = (i & 7) * 4;
      int gn = n0 + n;
      float4 f = make_float4(0.f, 0.f, 0.f, 0.f);
      if (gn < CO) f = *(const float4*)(w + (size_t)gn * K + k0 + kk0);
      int sub = n >> 4, nn = n & 15;
      int lane_d = (kk0 >> 4) * 16 + nn;
      int e = kk0 & 15;
      uint2 p; p.x = pk2(f.x, f.y); p.y = pk2(f.z, f.w);
      *(uint2*)&WsF[sub][lane_d][e] = p;
    }
    __syncthreads();

    Frag a;
    {
      const uint4* p = (const uint4*)&AsF[wave][lane][0];
      a.q[0] = p[0]; a.q[1] = p[1];
    }
#pragma unroll
    for (int j = 0; j < 4; ++j) {
      Frag bm;
      const uint4* p = (const uint4*)&WsF[j][lane][0];
      bm.q[0] = p[0]; bm.q[1] = p[1];
      acc[j] = __builtin_amdgcn_wmma_f32_16x16x32_bf16(
          false, a.v, false, bm.v, (short)0, acc[j], false, false);
    }
    __syncthreads();
  }

  int rbase = t0 + wave * 16 + hi * 8;
#pragma unroll
  for (int j = 0; j < 4; ++j) {
    int col = n0 + j * 16 + lo;
#pragma unroll
    for (int r = 0; r < 8; ++r) {
      int row = rbase + r;
      if (row < L && col < CO) {
        float v = acc[j][r] + bias[col];
        out[((size_t)b * CO + col) * L + row] = gelu_exact(v);
      }
    }
  }
}

// =====================================================================
// Flash attention, diag-masked, per (b,h). 4 waves/block, 64 queries/block,
// each wave owns 16 queries. K/V streamed through LDS in 64-key tiles,
// all tiles stored pre-swizzled in WMMA fragment order.
// q,k,v,o layout: (B*S, D) rows, head h occupies cols [h*32, h*32+32).
// =====================================================================
__global__ void __launch_bounds__(128) attention_kernel(
    const float* __restrict__ q, const float* __restrict__ k,
    const float* __restrict__ v, float* __restrict__ o,
    int S, int D, float scale)
{
  const int b    = blockIdx.z;
  const int h    = blockIdx.y;
  const int wave = threadIdx.x >> 5;
  const int lane = threadIdx.x & 31;
  const int hi   = (lane >= 16) ? 1 : 0;
  const int lo   = lane & 15;
  const int qrow0 = blockIdx.x * 64 + wave * 16;
  const size_t baseoff = (size_t)b * S * D + (size_t)h * HDIM;

  __shared__ __align__(16) unsigned short KsF[4][32][16];      // B-frags: K^T per 16-key subtile
  __shared__ __align__(16) unsigned short VsF[2][2][32][16];   // B-frags: V per (kchunk, hd-subtile)
  __shared__ __align__(16) unsigned short PsF[4][2][32][16];   // A-frags: P per (wave, kchunk)

  Frag qa;
  {
    int s = qrow0 + lo;
    int ab = hi * 8;
#pragma unroll
    for (int e = 0; e < 16; ++e) {
      int kk = 2 * ab + e;
      float val = (s < S) ? q[baseoff + (size_t)s * D + kk] : 0.0f;
      qa.u[e] = f2bf(val);
    }
  }

  f32x8 oacc[2];
  float mrun[8], lrun[8];
#pragma unroll
  for (int r = 0; r < 8; ++r) {
    mrun[r] = -1e30f; lrun[r] = 0.0f;
    oacc[0][r] = 0.0f; oacc[1][r] = 0.0f;
  }

  const int nkt = (S + 63) / 64;
  for (int kt = 0; kt < nkt; ++kt) {
    int kb0 = kt * 64;
    __syncthreads();
    // K tile: 64 keys x 32 hd, float4 over hd
    for (int i = threadIdx.x; i < 512; i += 128) {
      int r = i >> 3, c0 = (i & 7) * 4;
      int s = kb0 + r;
      float4 f = make_float4(0.f, 0.f, 0.f, 0.f);
      if (s < S) f = *(const float4*)(k + baseoff + (size_t)s * D + c0);
      int sub = r >> 4, n = r & 15;
      int lane_d = (c0 >> 4) * 16 + n;
      int e = c0 & 15;
      uint2 p; p.x = pk2(f.x, f.y); p.y = pk2(f.z, f.w);
      *(uint2*)&KsF[sub][lane_d][e] = p;
    }
    // V tile: 64 keys x 32 hd, 4 consecutive keys per chunk (keys are B-rows)
    for (int i = threadIdx.x; i < 512; i += 128) {
      int c = i & 31;          // hd
      int r0 = (i >> 5) * 4;   // key group
      float fv[4];
#pragma unroll
      for (int z = 0; z < 4; ++z) {
        int s = kb0 + r0 + z;
        fv[z] = (s < S) ? v[baseoff + (size_t)s * D + c] : 0.0f;
      }
      int kc = r0 >> 5;
      int kkl = r0 & 31;
      int lane_d = (kkl >> 4) * 16 + (c & 15);
      int e = kkl & 15;
      int st = c >> 4;
      uint2 p;
      p.x = pk2(fv[0], fv[1]);
      p.y = pk2(fv[2], fv[3]);
      *(uint2*)&VsF[kc][st][lane_d][e] = p;
    }
    __syncthreads();

    // scores: Q (16x32) x K^T (32x64) -> 4 WMMAs
    f32x8 sacc[4];
#pragma unroll
    for (int j = 0; j < 4; ++j) {
#pragma unroll
      for (int r = 0; r < 8; ++r) sacc[j][r] = 0.0f;
      Frag kf;
      const uint4* p = (const uint4*)&KsF[j][lane][0];
      kf.q[0] = p[0]; kf.q[1] = p[1];
      sacc[j] = __builtin_amdgcn_wmma_f32_16x16x32_bf16(
          false, qa.v, false, kf.v, (short)0, sacc[j], false, false);
    }

    // scale + mask + online softmax
    float sv[4][8];
    float pm[8];
#pragma unroll
    for (int r = 0; r < 8; ++r) pm[r] = -3e38f;
#pragma unroll
    for (int j = 0; j < 4; ++j) {
      int kcol = kb0 + j * 16 + lo;
#pragma unroll
      for (int r = 0; r < 8; ++r) {
        int qr = qrow0 + hi * 8 + r;
        float val = sacc[j][r] * scale;
        if (kcol >= S || qr >= S || kcol == qr) val = -3e38f;
        sv[j][r] = val;
        pm[r] = fmaxf(pm[r], val);
      }
    }
    const int m_ = hi * 8;  // row base within 16-row tile
#pragma unroll
    for (int r = 0; r < 8; ++r) {
      float mx = pm[r];
      mx = fmaxf(mx, __shfl_xor(mx, 1));
      mx = fmaxf(mx, __shfl_xor(mx, 2));
      mx = fmaxf(mx, __shfl_xor(mx, 4));
      mx = fmaxf(mx, __shfl_xor(mx, 8));
      float mnew = fmaxf(mrun[r], mx);
      float corr = expf(mrun[r] - mnew);
      float lsum = 0.0f;
#pragma unroll
      for (int j = 0; j < 4; ++j) {
        float p = expf(sv[j][r] - mnew);
        sv[j][r] = p;
        lsum += p;
      }
      lsum += __shfl_xor(lsum, 1);
      lsum += __shfl_xor(lsum, 2);
      lsum += __shfl_xor(lsum, 4);
      lsum += __shfl_xor(lsum, 8);
      lrun[r] = lrun[r] * corr + lsum;
      mrun[r] = mnew;
      oacc[0][r] *= corr;
      oacc[1][r] *= corr;
      // write P row into per-wave A-fragment layout
#pragma unroll
      for (int j = 0; j < 4; ++j) {
        int col = j * 16 + lo;
        int kc = col >> 5, kl = col & 31;
        int dl = (kl >> 4) * 16 + (m_ + r);
        int e = kl & 15;
        PsF[wave][kc][dl][e] = f2bf(sv[j][r]);
      }
    }

    // O += P (16x64) x V (64x32): 2 k-chunks x 2 output subtiles
#pragma unroll
    for (int kc = 0; kc < 2; ++kc) {
      Frag pa;
      const uint4* pp = (const uint4*)&PsF[wave][kc][lane][0];
      pa.q[0] = pp[0]; pa.q[1] = pp[1];
#pragma unroll
      for (int st = 0; st < 2; ++st) {
        Frag vb;
        const uint4* vp = (const uint4*)&VsF[kc][st][lane][0];
        vb.q[0] = vp[0]; vb.q[1] = vp[1];
        oacc[st] = __builtin_amdgcn_wmma_f32_16x16x32_bf16(
            false, pa.v, false, vb.v, (short)0, oacc[st], false, false);
      }
    }
  }

#pragma unroll
  for (int st = 0; st < 2; ++st) {
#pragma unroll
    for (int r = 0; r < 8; ++r) {
      int s = qrow0 + hi * 8 + r;
      if (s < S) {
        o[baseoff + (size_t)s * D + st * 16 + lo] = oacc[st][r] / lrun[r];
      }
    }
  }
}

// =====================================================================
// Small scalar kernels
// =====================================================================
__global__ void __launch_bounds__(256) instnorm_kernel(
    const float* __restrict__ x, float* __restrict__ load_norm,
    float* __restrict__ imean, float* __restrict__ istd, int L)
{
  int b = blockIdx.x, tid = threadIdx.x;
  float s = 0.0f, s2 = 0.0f;
  for (int t = tid; t < L; t += 256) {
    float v = x[((size_t)b * L + t) * 9];
    s += v; s2 += v * v;
  }
  __shared__ float sh[256], sh2[256];
  sh[tid] = s; sh2[tid] = s2; __syncthreads();
  for (int st = 128; st > 0; st >>= 1) {
    if (tid < st) { sh[tid] += sh[tid + st]; sh2[tid] += sh2[tid + st]; }
    __syncthreads();
  }
  __shared__ float mS, sS;
  if (tid == 0) {
    float m = sh[0] / L;
    float var = sh2[0] / L - m * m;
    float sd = sqrtf(var + 1e-6f);
    imean[b] = m; istd[b] = sd; mS = m; sS = sd;
  }
  __syncthreads();
  float m = mS, sd = sS;
  for (int t = tid; t < L; t += 256)
    load_norm[(size_t)b * L + t] = (x[((size_t)b * L + t) * 9] - m) / sd;
}

__global__ void conv1_gelu_kernel(
    const float* __restrict__ in, const float* __restrict__ w,
    const float* __restrict__ bias, float* __restrict__ out, int L, int total)
{
  int idx = blockIdx.x * blockDim.x + threadIdx.x;
  if (idx >= total) return;
  int t = idx % L;
  int c = (idx / L) % CMAIN;
  int b = idx / (L * CMAIN);
  float a = bias[c];
#pragma unroll
  for (int ks = 0; ks < 3; ++ks) {
    int tt = t + (ks - 1);
    if (tt >= 0 && tt < L) a += w[c * 3 + ks] * in[(size_t)b * L + tt];
  }
  out[idx] = gelu_exact(a);
}

__global__ void __launch_bounds__(256) bn_stats_kernel(
    const float* __restrict__ h, float* __restrict__ mean,
    float* __restrict__ rstd, int C, int L)
{
  int c = blockIdx.x, tid = threadIdx.x;
  float s = 0.0f, s2 = 0.0f;
  int n = NB * L;
  for (int i = tid; i < n; i += 256) {
    int b = i / L, t = i % L;
    float v = h[((size_t)b * C + c) * L + t];
    s += v; s2 += v * v;
  }
  __shared__ float sh[256], sh2[256];
  sh[tid] = s; sh2[tid] = s2; __syncthreads();
  for (int st = 128; st > 0; st >>= 1) {
    if (tid < st) { sh[tid] += sh[tid + st]; sh2[tid] += sh2[tid + st]; }
    __syncthreads();
  }
  if (tid == 0) {
    float m = sh[0] / n;
    float var = sh2[0] / n - m * m;
    mean[c] = m; rstd[c] = rsqrtf(var + 1e-5f);
  }
}

__global__ void bn_res_kernel(
    const float* __restrict__ xres, const float* __restrict__ h,
    const float* __restrict__ mean, const float* __restrict__ rstd,
    const float* __restrict__ g, const float* __restrict__ beta,
    float* __restrict__ out, int C, int L, int total, int resBroadcast)
{
  int idx = blockIdx.x * blockDim.x + threadIdx.x;
  if (idx >= total) return;
  int t = idx % L;
  int c = (idx / L) % C;
  int b = idx / (L * C);
  float res = resBroadcast ? xres[(size_t)b * L + t] : xres[idx];
  out[idx] = res + (h[idx] - mean[c]) * rstd[c] * g[c] + beta[c];
}

__global__ void proj_emb_kernel(
    const float* __restrict__ x, const float* __restrict__ w,
    const float* __restrict__ bias, float* __restrict__ out, int L, int total)
{
  int idx = blockIdx.x * blockDim.x + threadIdx.x;
  if (idx >= total) return;
  int t = idx % L;
  int c = (idx / L) % CEXO;
  int b = idx / (L * CEXO);
  float a = bias[c];
#pragma unroll
  for (int ci = 0; ci < CEMB; ++ci)
    a += w[c * CEMB + ci] * x[((size_t)b * L + t) * 9 + 1 + ci];
  out[idx] = a;
}

__global__ void assemble_kernel(
    const float* __restrict__ conv, const float* __restrict__ enc,
    const float* __restrict__ sw, const float* __restrict__ sb,
    const float* __restrict__ imean, const float* __restrict__ istd,
    float* __restrict__ xtf, int L, int S, int D, int total)
{
  int idx = blockIdx.x * blockDim.x + threadIdx.x;
  if (idx >= total) return;
  int d = idx % D;
  int s = (idx / D) % S;
  int b = idx / (D * S);
  float v;
  if (s < L) {
    v = (d < CMAIN) ? conv[((size_t)b * CMAIN + d) * L + s]
                    : enc[((size_t)b * CEXO + (d - CMAIN)) * L + s];
  } else {
    v = sb[d] + sw[d * 2 + 0] * imean[b] + sw[d * 2 + 1] * istd[b];
  }
  xtf[idx] = v;
}

__global__ void __launch_bounds__(256) layernorm_kernel(
    const float* __restrict__ x, const float* __restrict__ g,
    const float* __restrict__ b_, float* __restrict__ y, int D)
{
  int row = blockIdx.x, tid = threadIdx.x;
  float v = x[(size_t)row * D + tid];
  __shared__ float s1[256], s2[256];
  s1[tid] = v; s2[tid] = v * v; __syncthreads();
  for (int st = 128; st > 0; st >>= 1) {
    if (tid < st) { s1[tid] += s1[tid + st]; s2[tid] += s2[tid + st]; }
    __syncthreads();
  }
  float m = s1[0] / D;
  float var = s2[0] / D - m * m;
  float r = rsqrtf(var + 1e-5f);
  y[(size_t)row * D + tid] = (v - m) * r * g[tid] + b_[tid];
}

__global__ void __launch_bounds__(128) heads_kernel(
    const float* __restrict__ xn,
    const float* __restrict__ cw1, const float* __restrict__ cb1,
    const float* __restrict__ cw2, const float* __restrict__ cb2,
    const float* __restrict__ hw1, const float* __restrict__ hb1,
    const float* __restrict__ hw2, const float* __restrict__ hb2,
    const float* __restrict__ imean, const float* __restrict__ istd,
    float* __restrict__ out, int S, int D, int mid)
{
  int b = blockIdx.x, tid = threadIdx.x;
  __shared__ float feat[256];
  for (int d = tid; d < D; d += 128)
    feat[d] = xn[((size_t)b * S + mid) * D + d];
  __syncthreads();

  float a = cb1[tid], h2 = hb1[tid];
  for (int d = 0; d < D; ++d) {
    float f = feat[d];
    a  += cw1[tid * D + d] * f;
    h2 += hw1[tid * D + d] * f;
  }
  a = fmaxf(a, 0.0f);
  h2 = fmaxf(h2, 0.0f);
  __shared__ float r1[128], r2[128];
  r1[tid] = a * cw2[tid];
  r2[tid] = h2 * hw2[tid];
  __syncthreads();
  for (int st = 64; st > 0; st >>= 1) {
    if (tid < st) { r1[tid] += r1[tid + st]; r2[tid] += r2[tid + st]; }
    __syncthreads();
  }
  if (tid == 0) {
    float sl = r1[0] + cb2[0];
    float pr = r2[0] + hb2[0];
    float denorm = fmaxf(pr * istd[b] + imean[b], 0.0f);
    float prob = 1.0f / (1.0f + expf(-sl));
    out[b]      = denorm * prob;   // power_final
    out[8 + b]  = sl;              // state_logits
    out[16 + b] = denorm;          // power_denorm
  }
}

// =====================================================================
// host-side launcher
// =====================================================================
static void launch_gemm(const float* A, const float* W, const float* bias,
                        const float* resid, float* C, int M, int N, int K,
                        int act, hipStream_t s)
{
  dim3 g((M + 63) / 64, N / 64), b(128);
  gemm_wmma_kernel<<<g, b, 0, s>>>(A, W, bias, resid, C, M, N, K, act);
}

extern "C" void kernel_launch(void* const* d_in, const int* in_sizes, int n_in,
                              void* d_out, int out_size, void* d_ws, size_t ws_size,
                              hipStream_t stream) {
  (void)in_sizes; (void)out_size; (void)ws_size;
  if (n_in < 67) return;

  // ---- parameter mapping (setup_inputs dict insertion order, recursive) ----
  const float* X = (const float*)d_in[0];
  int ii = 1;
  struct ConvP { const float *w, *b, *g, *beta; } cp[4];
  for (int u = 0; u < 4; ++u) {
    cp[u].w    = (const float*)d_in[ii++];
    cp[u].b    = (const float*)d_in[ii++];
    cp[u].g    = (const float*)d_in[ii++];
    cp[u].beta = (const float*)d_in[ii++];
  }
  const float* proj_w  = (const float*)d_in[ii++];
  const float* proj_b  = (const float*)d_in[ii++];
  const float* stats_w = (const float*)d_in[ii++];
  const float* stats_b = (const float*)d_in[ii++];
  struct LayerP { const float *ln1_g,*ln1_b,*wq,*wk,*wv,*wo,*ln2_g,*ln2_b,*w1,*b1,*w2,*b2; } lp[3];
  for (int l = 0; l < 3; ++l) {
    lp[l].ln1_g = (const float*)d_in[ii++]; lp[l].ln1_b = (const float*)d_in[ii++];
    lp[l].wq = (const float*)d_in[ii++]; lp[l].wk = (const float*)d_in[ii++];
    lp[l].wv = (const float*)d_in[ii++]; lp[l].wo = (const float*)d_in[ii++];
    lp[l].ln2_g = (const float*)d_in[ii++]; lp[l].ln2_b = (const float*)d_in[ii++];
    lp[l].w1 = (const float*)d_in[ii++]; lp[l].b1 = (const float*)d_in[ii++];
    lp[l].w2 = (const float*)d_in[ii++]; lp[l].b2 = (const float*)d_in[ii++];
  }
  const float* final_g = (const float*)d_in[ii++];
  const float* final_b = (const float*)d_in[ii++];
  const float* cls_w1  = (const float*)d_in[ii++];
  const float* cls_b1  = (const float*)d_in[ii++];
  const float* cls_w2  = (const float*)d_in[ii++];
  const float* cls_b2  = (const float*)d_in[ii++];
  const float* head_w1 = (const float*)d_in[ii++];
  const float* head_b1 = (const float*)d_in[ii++];
  const float* head_w2 = (const float*)d_in[ii++];
  const float* head_b2 = (const float*)d_in[ii++];

  // ---- workspace layout (floats); lifetime-based overlap ----
  float* ws = (float*)d_ws;
  const size_t CBUF = (size_t)NB * CMAIN * LL;      // 3,145,728
  const size_t XBUF = (size_t)NB * SS * DMODEL;     // 4,196,352
  float* imean = ws + 0;
  float* istd  = ws + 8;
  float* chm   = ws + 16;
  float* chr   = ws + 208;
  float* lnorm = ws + 1024;                         // B*L
  float* C1    = ws + 17408;
  float* C2    = C1 + CBUF;
  float* C3    = C2 + CBUF;
  float* ENC   = C3 + CBUF;                         // B*64*L
  float* XTF   = ENC + (size_t)NB * CEXO * LL;
  float* XN    = C1;                                // alias: dead conv buffers
  float* Q     = XTF + XBUF;
  float* Kb    = Q + XBUF;
  float* Vb    = Kb + XBUF;
  float* ATT   = Vb + XBUF;
  float* FFNM  = Q;                                 // alias: q/k/v/att dead during FFN

  const int convTotal = (int)CBUF;
  const dim3 convGrid(LL / 64, CMAIN / 64, NB);

  // ---- stage 1: instance norm ----
  instnorm_kernel<<<NB, 256, 0, stream>>>(X, lnorm, imean, istd, LL);

  // ---- stage 2: dilated conv residual stack ----
  // unit 0 (cin=1, dil=1): scalar conv; residual broadcast of load_norm
  conv1_gelu_kernel<<<(convTotal + 255) / 256, 256, 0, stream>>>(
      lnorm, cp[0].w, cp[0].b, C3, LL, convTotal);
  bn_stats_kernel<<<CMAIN, 256, 0, stream>>>(C3, chm, chr, CMAIN, LL);
  bn_res_kernel<<<(convTotal + 255) / 256, 256, 0, stream>>>(
      lnorm, C3, chm, chr, cp[0].g, cp[0].beta, C1, CMAIN, LL, convTotal, 1);
  // units 1..3 (cin=192, dil=2,4,8): WMMA implicit-GEMM conv, ping-pong C1/C2
  const int dils[3] = {2, 4, 8};
  float* pin = C1; float* pout = C2;
  for (int u = 1; u <= 3; ++u) {
    conv_wmma_kernel<<<convGrid, 128, 0, stream>>>(
        pin, cp[u].w, cp[u].b, C3, CMAIN, CMAIN, LL, dils[u - 1]);
    bn_stats_kernel<<<CMAIN, 256, 0, stream>>>(C3, chm, chr, CMAIN, LL);
    bn_res_kernel<<<(convTotal + 255) / 256, 256, 0, stream>>>(
        pin, C3, chm, chr, cp[u].g, cp[u].beta, pout, CMAIN, LL, convTotal, 0);
    float* t = pin; pin = pout; pout = t;
  }
  float* convFinal = pin;  // after 3 swaps: C2

  // ---- stage 3: encoding projection + token assembly ----
  const int projTotal = NB * CEXO * LL;
  proj_emb_kernel<<<(projTotal + 255) / 256, 256, 0, stream>>>(
      X, proj_w, proj_b, ENC, LL, projTotal);
  const int xtfTotal = (int)XBUF;
  assemble_kernel<<<(xtfTotal + 255) / 256, 256, 0, stream>>>(
      convFinal, ENC, stats_w, stats_b, imean, istd, XTF, LL, SS, DMODEL, xtfTotal);

  // ---- stage 4: transformer encoder layers ----
  const float scale = 1.0f / sqrtf((float)HDIM);
  for (int l = 0; l < 3; ++l) {
    layernorm_kernel<<<BSROWS, 256, 0, stream>>>(XTF, lp[l].ln1_g, lp[l].ln1_b, XN, DMODEL);
    launch_gemm(XN, lp[l].wq, nullptr, nullptr, Q,  BSROWS, DMODEL, DMODEL, 0, stream);
    launch_gemm(XN, lp[l].wk, nullptr, nullptr, Kb, BSROWS, DMODEL, DMODEL, 0, stream);
    launch_gemm(XN, lp[l].wv, nullptr, nullptr, Vb, BSROWS, DMODEL, DMODEL, 0, stream);
    dim3 ag((SS + 63) / 64, NH, NB);
    attention_kernel<<<ag, 128, 0, stream>>>(Q, Kb, Vb, ATT, SS, DMODEL, scale);
    launch_gemm(ATT, lp[l].wo, nullptr, XTF, XTF, BSROWS, DMODEL, DMODEL, 0, stream);
    layernorm_kernel<<<BSROWS, 256, 0, stream>>>(XTF, lp[l].ln2_g, lp[l].ln2_b, XN, DMODEL);
    launch_gemm(XN, lp[l].w1, lp[l].b1, nullptr, FFNM, BSROWS, DFF, DMODEL, 1, stream);
    launch_gemm(FFNM, lp[l].w2, lp[l].b2, XTF, XTF, BSROWS, DMODEL, DFF, 0, stream);
  }

  // ---- stage 5: final LN + heads ----
  layernorm_kernel<<<BSROWS, 256, 0, stream>>>(XTF, final_g, final_b, XN, DMODEL);
  heads_kernel<<<NB, 128, 0, stream>>>(
      XN, cls_w1, cls_b1, cls_w2, cls_b2,
      head_w1, head_b1, head_w2, head_b2,
      imean, istd, (float*)d_out, SS, DMODEL, LL / 2);
}
